// VectorQuantizer_66383014527027
// MI455X (gfx1250) — matrix operand order repaired and checked by reading, not compiled, not run on hardware
//
#include <hip/hip_runtime.h>

typedef __attribute__((ext_vector_type(16))) __bf16 v16bf;
typedef __attribute__((ext_vector_type(8)))  float  v8f;

#define N_TOTAL    262144
#define NUM_CODES  1024
#define CODE_DIM   64
#define BLOCK_ROWS 256          // rows per workgroup (8 waves x 32 rows)
#define NBLOCKS    (N_TOTAL / BLOCK_ROWS)   // 1024
#define HALF_CODES 512          // codebook staged in two halves
#define SLOTS      64           // 32 tiles x 2 K-fragments per half

__global__ __launch_bounds__(256, 1)
void VectorQuantizer_66383014527027_kernel(const float* __restrict__ x,
                                           const float* __restrict__ cb,
                                           float* __restrict__ out)
{
    // B fragments, pre-swizzled: [slot][comp(hi/lo)][lane][16 bf16]
    __shared__ __bf16 sB[SLOTS * 2 * 512];       // 128 KB
    __shared__ float  sCsq[NUM_CODES];           // 4 KB
    __shared__ int    sIdx[BLOCK_ROWS];          // 1 KB

    const int tid  = threadIdx.x;
    const int lane = tid & 31;
    const int wave = tid >> 5;                   // 0..7
    const int nloc = lane & 15;                  // column-in-tile / row-in-tile

    // ---- per-code squared norms (fp32 exact), once per block ----
    for (int c = tid; c < NUM_CODES; c += 256) {
        const float4* r = (const float4*)(cb + c * CODE_DIM);
        float s = 0.f;
#pragma unroll
        for (int q = 0; q < 16; ++q) {
            float4 v = r[q];
            s += v.x * v.x + v.y * v.y + v.z * v.z + v.w * v.w;
        }
        sCsq[c] = s;
    }

    // ---- load this wave's 32 input rows as two A-tile fragment sets ----
    // A 16x32 layout (matches B staging): lanes<16 hold K {0..15}, lanes>=16 hold K {16..31}
    const int rowbase = blockIdx.x * BLOCK_ROWS + wave * 32;
    const int kb0 = (lane < 16) ? 0 : 16;
    v16bf ah[2][2], al[2][2];
#pragma unroll
    for (int mt = 0; mt < 2; ++mt) {
        const float* rp = x + (size_t)(rowbase + mt * 16 + nloc) * CODE_DIM;
#pragma unroll
        for (int f = 0; f < 2; ++f) {
            float tmp[16];
            float4 c0 = *(const float4*)(rp + f * 32 + kb0 + 0);
            float4 c1 = *(const float4*)(rp + f * 32 + kb0 + 4);
            float4 c2 = *(const float4*)(rp + f * 32 + kb0 + 8);
            float4 c3 = *(const float4*)(rp + f * 32 + kb0 + 12);
            tmp[0]=c0.x;  tmp[1]=c0.y;  tmp[2]=c0.z;  tmp[3]=c0.w;
            tmp[4]=c1.x;  tmp[5]=c1.y;  tmp[6]=c1.z;  tmp[7]=c1.w;
            tmp[8]=c2.x;  tmp[9]=c2.y;  tmp[10]=c2.z; tmp[11]=c2.w;
            tmp[12]=c3.x; tmp[13]=c3.y; tmp[14]=c3.z; tmp[15]=c3.w;
            v16bf h, l;
#pragma unroll
            for (int j = 0; j < 16; ++j) {
                __bf16 hb = (__bf16)tmp[j];
                h[j] = hb;
                l[j] = (__bf16)(tmp[j] - (float)hb);
            }
            ah[mt][f] = h;
            al[mt][f] = l;
        }
    }

    float best[2][8];
    int   bidx[2][8];
#pragma unroll
    for (int mt = 0; mt < 2; ++mt)
#pragma unroll
        for (int i = 0; i < 8; ++i) { best[mt][i] = 3.4e38f; bidx[mt][i] = 0; }

    // ---- two codebook halves: stage bf16 hi/lo B-fragments, then 32 code tiles ----
    for (int h = 0; h < 2; ++h) {
        if (h) __syncthreads();   // all waves done reading previous half
        // stage: slot = local_tile*2 + frag; lane n<16 -> code row 16t+n, K 32f..+15
        //        lane n>=16 -> same row, K 32f+16..+31  (contiguous K order, same as A)
        for (int slot = wave; slot < SLOTS; slot += 8) {
            int t  = slot >> 1, f = slot & 1;
            int row = h * HALF_CODES + t * 16 + nloc;
            int kb  = f * 32 + ((lane < 16) ? 0 : 16);
            const float4* src = (const float4*)(cb + row * CODE_DIM + kb);
            float tmp[16];
#pragma unroll
            for (int q = 0; q < 4; ++q) {
                float4 v = src[q];
                tmp[q*4+0]=v.x; tmp[q*4+1]=v.y; tmp[q*4+2]=v.z; tmp[q*4+3]=v.w;
            }
            v16bf hv, lv;
#pragma unroll
            for (int j = 0; j < 16; ++j) {
                __bf16 hb = (__bf16)tmp[j];
                hv[j] = hb;
                lv[j] = (__bf16)(tmp[j] - (float)hb);
            }
            *(v16bf*)&sB[(slot * 2 + 0) * 512 + lane * 16] = hv;
            *(v16bf*)&sB[(slot * 2 + 1) * 512 + lane * 16] = lv;
        }
        __syncthreads();

        for (int t = 0; t < 32; ++t) {
            v16bf bh0 = *(const v16bf*)&sB[((t*2+0)*2 + 0) * 512 + lane * 16];
            v16bf bl0 = *(const v16bf*)&sB[((t*2+0)*2 + 1) * 512 + lane * 16];
            v16bf bh1 = *(const v16bf*)&sB[((t*2+1)*2 + 0) * 512 + lane * 16];
            v16bf bl1 = *(const v16bf*)&sB[((t*2+1)*2 + 1) * 512 + lane * 16];
            const int n = h * HALF_CODES + t * 16 + nloc;
            const float cs = sCsq[n];
#pragma unroll
            for (int mt = 0; mt < 2; ++mt) {
                v8f acc = {};
                // hi*hi + hi*lo + lo*hi  (fp32 accumulate) over K=64 (two K=32 frags)
                acc = __builtin_amdgcn_wmma_f32_16x16x32_bf16(false, ah[mt][0], false, bh0, (short)0, acc, false, false);
                acc = __builtin_amdgcn_wmma_f32_16x16x32_bf16(false, ah[mt][1], false, bh1, (short)0, acc, false, false);
                acc = __builtin_amdgcn_wmma_f32_16x16x32_bf16(false, ah[mt][0], false, bl0, (short)0, acc, false, false);
                acc = __builtin_amdgcn_wmma_f32_16x16x32_bf16(false, ah[mt][1], false, bl1, (short)0, acc, false, false);
                acc = __builtin_amdgcn_wmma_f32_16x16x32_bf16(false, al[mt][0], false, bh0, (short)0, acc, false, false);
                acc = __builtin_amdgcn_wmma_f32_16x16x32_bf16(false, al[mt][1], false, bh1, (short)0, acc, false, false);
#pragma unroll
                for (int i = 0; i < 8; ++i) {
                    float d = fmaf(-2.0f, acc[i], cs);   // ||c||^2 - 2 x.c
                    if (d < best[mt][i]) { best[mt][i] = d; bidx[mt][i] = n; }
                }
            }
        }
    }

    // ---- cross-lane argmin within each 16-lane group (N dimension of the tile) ----
#pragma unroll
    for (int mt = 0; mt < 2; ++mt) {
#pragma unroll
        for (int i = 0; i < 8; ++i) {
            float v  = best[mt][i];
            int   ix = bidx[mt][i];
#pragma unroll
            for (int m2 = 8; m2 >= 1; m2 >>= 1) {
                float ov = __shfl_xor(v, m2, 32);
                int   oi = __shfl_xor(ix, m2, 32);
                if (ov < v || (ov == v && oi < ix)) { v = ov; ix = oi; }
            }
            if (nloc == 0) {
                // lane 0 -> rows M=i of tile mt, lane 16 -> rows M=8+i
                int rl = wave * 32 + mt * 16 + ((lane >> 4) << 3) + i;
                sIdx[rl] = ix;
            }
        }
    }
    __syncthreads();

    // ---- cooperative gather: copy winning fp32 codebook rows to output ----
    const int c4 = tid & 15;         // 16 x float4 = 64 floats per row
    const int rsub = tid >> 4;       // 16 rows per pass
    for (int pass = 0; pass < 16; ++pass) {
        int rl  = pass * 16 + rsub;
        int idx = sIdx[rl];
        float4 v = ((const float4*)(cb + idx * CODE_DIM))[c4];
        ((float4*)(out + (size_t)(blockIdx.x * BLOCK_ROWS + rl) * CODE_DIM))[c4] = v;
    }
}

extern "C" void kernel_launch(void* const* d_in, const int* in_sizes, int n_in,
                              void* d_out, int out_size, void* d_ws, size_t ws_size,
                              hipStream_t stream) {
    (void)in_sizes; (void)n_in; (void)out_size; (void)d_ws; (void)ws_size;
    const float* x  = (const float*)d_in[0];
    const float* cb = (const float*)d_in[1];
    float* out = (float*)d_out;
    VectorQuantizer_66383014527027_kernel<<<dim3(NBLOCKS), dim3(256), 0, stream>>>(x, cb, out);
}
